// GraphNeuralNetwork_30605936951830
// MI455X (gfx1250) — matrix operand order repaired and checked by reading, not compiled, not run on hardware
//
#include <hip/hip_runtime.h>
#include <hip/hip_bf16.h>

// ---------------------------------------------------------------------------
// GNN (GatedGCN-style MPNN): 3 layers of
//   e' = MLP([h[src], h[dst], e])  (192->64->64->64->64, ReLU all)
//   agg = segment_sum(e', dst)
//   h' = MLP([agg, h])             (128->64->64->64->64, ReLU all)
// Strategy: bf16 WMMA (16x16x32, f32 acc). Each wave processes TWO 16-row
// tiles: B fragments are loaded once per pair (halved weight traffic) and the
// two accumulator chains interleave so WMMA->WMMA hazard slots are filled by
// independent work. Weights pre-packed into exact B-fragment order;
// activations staged per-wave in LDS (DS in-order, s_wait_dscnt fence).
// ---------------------------------------------------------------------------

typedef __attribute__((ext_vector_type(16))) __bf16 v16bf;
typedef __attribute__((ext_vector_type(8)))  __bf16 v8bf;
typedef __attribute__((ext_vector_type(8)))  float  v8f;

#define LDS_STRIDE 72   // bf16 elems per row; 144B rows keep 16B alignment

__device__ __forceinline__ v8f wmma_bf16(v16bf a, v16bf b, v8f c) {
  return __builtin_amdgcn_wmma_f32_16x16x32_bf16(
      /*neg_a=*/false, a, /*neg_b=*/false, b,
      /*c_mod=*/(short)0, c, /*reuse_a=*/false, /*reuse_b=*/false);
}

__device__ __forceinline__ void lds_fence() {
  // DS is in-order per wave; this is a compiler barrier + explicit wait.
  asm volatile("s_wait_dscnt 0" ::: "memory");
}

__device__ __forceinline__ void init_bias(v8f acc[4], const float* __restrict__ bias, int col) {
#pragma unroll
  for (int ct = 0; ct < 4; ct++) {
    float bv = bias[ct * 16 + col];
#pragma unroll
    for (int r = 0; r < 8; r++) acc[ct][r] = bv;
  }
}

// Pair matmul: two 16x64 LDS bf16 tiles x 64x64 packed weights + bias.
__device__ __forceinline__ void lds_mm64_pair(const __bf16* __restrict__ t0,
                                              const __bf16* __restrict__ t1,
                                              const __bf16* __restrict__ wp,
                                              const float* __restrict__ bias,
                                              int lane, v8f acc0[4], v8f acc1[4]) {
  const int row = lane & 15;
  const int khalf = lane >> 4;
  init_bias(acc0, bias, row);
  init_bias(acc1, bias, row);
#pragma unroll
  for (int kb = 0; kb < 2; kb++) {
    v16bf a0, a1;
#pragma unroll
    for (int g = 0; g < 2; g++) {
      const int o = row * LDS_STRIDE + kb * 32 + khalf * 8 + g * 16;
      v8bf x0 = *(const v8bf*)(t0 + o);
      v8bf x1 = *(const v8bf*)(t1 + o);
#pragma unroll
      for (int j = 0; j < 8; j++) { a0[g * 8 + j] = x0[j]; a1[g * 8 + j] = x1[j]; }
    }
    v16bf bf[4];
#pragma unroll
    for (int ct = 0; ct < 4; ct++)
      bf[ct] = *(const v16bf*)(wp + ((size_t)(kb * 4 + ct) * 32 + lane) * 16);
#pragma unroll
    for (int ct = 0; ct < 4; ct++) {
      acc0[ct] = wmma_bf16(a0, bf[ct], acc0[ct]);
      acc1[ct] = wmma_bf16(a1, bf[ct], acc1[ct]);
    }
  }
}

// relu(acc) -> LDS tile (D layout: lane = col, VGPR r = row khalf*8+r)
__device__ __forceinline__ void relu_store_lds(__bf16* __restrict__ tile,
                                               const v8f acc[4], int lane) {
  const int c = lane & 15;
  const int h = lane >> 4;
#pragma unroll
  for (int ct = 0; ct < 4; ct++)
#pragma unroll
    for (int r = 0; r < 8; r++) {
      float v = acc[ct][r];
      v = v > 0.f ? v : 0.f;
      tile[(h * 8 + r) * LDS_STRIDE + ct * 16 + c] = (__bf16)v;
    }
}

// Load 8 consecutive f32, convert to bf16 into a[g*8 .. g*8+7]
__device__ __forceinline__ void cvt8(v16bf& a, int g, const float* __restrict__ p) {
  float4 x0 = *(const float4*)p;
  float4 x1 = *(const float4*)(p + 4);
  a[g * 8 + 0] = (__bf16)x0.x; a[g * 8 + 1] = (__bf16)x0.y;
  a[g * 8 + 2] = (__bf16)x0.z; a[g * 8 + 3] = (__bf16)x0.w;
  a[g * 8 + 4] = (__bf16)x1.x; a[g * 8 + 5] = (__bf16)x1.y;
  a[g * 8 + 6] = (__bf16)x1.z; a[g * 8 + 7] = (__bf16)x1.w;
}

// ---------------------------------------------------------------------------
// Weight pre-pack: f32 [Kin][64] -> bf16 in B-fragment order:
//   packed[((kb*4+ct)*32 + lane)*16 + elem] = W[K][c]
//   c = ct*16 + (lane&15);  K = kb*32 + (elem>>3)*16 + (lane>>4)*8 + (elem&7)
// ---------------------------------------------------------------------------
__global__ void pack_weight_kernel(const float* __restrict__ W, __bf16* __restrict__ out,
                                   int Kin) {
  int i = blockIdx.x * 256 + threadIdx.x;
  int total = Kin * 64;
  if (i >= total) return;
  int elem = i & 15;
  int lane = (i >> 4) & 31;
  int blk  = i >> 9;
  int ct = blk & 3;
  int kb = blk >> 2;
  int c = ct * 16 + (lane & 15);
  int K = kb * 32 + (elem >> 3) * 16 + (lane >> 4) * 8 + (elem & 7);
  out[i] = (__bf16)W[(size_t)K * 64 + c];
}

__global__ void zero_f32_kernel(float* __restrict__ p, int n) {
  int i = blockIdx.x * 256 + threadIdx.x;
  if (i < n) p[i] = 0.f;
}

// ---------------------------------------------------------------------------
// Edge kernel: per wave, TWO 16-edge tiles, A = [h[src] | h[dst] | e] (16x192),
// 4-layer MLP, write e_out and atomically scatter into agg[dst].
// ---------------------------------------------------------------------------
__global__ __launch_bounds__(256) void gnn_edge_kernel(
    const float* __restrict__ h, const float* __restrict__ e_in,
    float* __restrict__ e_out, float* __restrict__ agg,
    const int* __restrict__ src, const int* __restrict__ dst,
    const __bf16* __restrict__ w0, const __bf16* __restrict__ w1,
    const __bf16* __restrict__ w2, const __bf16* __restrict__ w3,
    const float* __restrict__ b0, const float* __restrict__ b1,
    const float* __restrict__ b2, const float* __restrict__ b3,
    int n_tiles) {
  __shared__ __align__(16) __bf16 lds_tiles[16][16 * LDS_STRIDE];
  __shared__ int lds_dsti[16][16];
  const int wave = threadIdx.x >> 5;
  const int lane = threadIdx.x & 31;
  const int t0 = (blockIdx.x * 8 + wave) * 2;
  if (t0 >= n_tiles) return;
  const bool has1 = (t0 + 1) < n_tiles;
  const int t1 = has1 ? (t0 + 1) : t0;  // clamp; duplicate work discarded
  __bf16* tile0 = &lds_tiles[wave * 2 + 0][0];
  __bf16* tile1 = &lds_tiles[wave * 2 + 1][0];
  const int row = lane & 15;
  const int khalf = lane >> 4;
  const int ei0 = t0 * 16 + row;
  const int ei1 = t1 * 16 + row;
  const int s0 = src[ei0], d0 = dst[ei0];
  const int s1 = src[ei1], d1 = dst[ei1];
  if (lane < 16) {
    lds_dsti[wave * 2 + 0][row] = d0;
    lds_dsti[wave * 2 + 1][row] = d1;
  }

  v8f acc0[4], acc1[4];
  init_bias(acc0, b0, row);
  init_bias(acc1, b0, row);
#pragma unroll
  for (int kb = 0; kb < 6; kb++) {
    v16bf a0, a1;
#pragma unroll
    for (int g = 0; g < 2; g++) {
      const int kbase = kb * 32 + g * 16;      // compile-time after unroll
      const int seg = kbase >> 6;              // 0: h[src], 1: h[dst], 2: e
      const int off = (kbase & 63) + khalf * 8;
      const float* p0 = (seg == 0) ? (h + (size_t)s0 * 64 + off)
                      : (seg == 1) ? (h + (size_t)d0 * 64 + off)
                                   : (e_in + (size_t)ei0 * 64 + off);
      const float* p1 = (seg == 0) ? (h + (size_t)s1 * 64 + off)
                      : (seg == 1) ? (h + (size_t)d1 * 64 + off)
                                   : (e_in + (size_t)ei1 * 64 + off);
      cvt8(a0, g, p0);
      cvt8(a1, g, p1);
    }
    v16bf bf[4];
#pragma unroll
    for (int ct = 0; ct < 4; ct++)
      bf[ct] = *(const v16bf*)(w0 + ((size_t)(kb * 4 + ct) * 32 + lane) * 16);
#pragma unroll
    for (int ct = 0; ct < 4; ct++) {
      acc0[ct] = wmma_bf16(a0, bf[ct], acc0[ct]);
      acc1[ct] = wmma_bf16(a1, bf[ct], acc1[ct]);
    }
  }
  relu_store_lds(tile0, acc0, lane);
  relu_store_lds(tile1, acc1, lane);
  lds_fence();
  lds_mm64_pair(tile0, tile1, w1, b1, lane, acc0, acc1);
  relu_store_lds(tile0, acc0, lane);
  relu_store_lds(tile1, acc1, lane);
  lds_fence();
  lds_mm64_pair(tile0, tile1, w2, b2, lane, acc0, acc1);
  relu_store_lds(tile0, acc0, lane);
  relu_store_lds(tile1, acc1, lane);
  lds_fence();
  lds_mm64_pair(tile0, tile1, w3, b3, lane, acc0, acc1);

  // final relu + writeback + scatter (segment_sum via f32 atomics)
#pragma unroll
  for (int ct = 0; ct < 4; ct++)
#pragma unroll
    for (int r = 0; r < 8; r++) {
      const int rr = khalf * 8 + r;
      const int col = ct * 16 + row;
      float v = acc0[ct][r];
      v = v > 0.f ? v : 0.f;
      e_out[(size_t)(t0 * 16 + rr) * 64 + col] = v;
      unsafeAtomicAdd(agg + (size_t)lds_dsti[wave * 2 + 0][rr] * 64 + col, v);
    }
  if (has1) {
#pragma unroll
    for (int ct = 0; ct < 4; ct++)
#pragma unroll
      for (int r = 0; r < 8; r++) {
        const int rr = khalf * 8 + r;
        const int col = ct * 16 + row;
        float v = acc1[ct][r];
        v = v > 0.f ? v : 0.f;
        e_out[(size_t)(t1 * 16 + rr) * 64 + col] = v;
        unsafeAtomicAdd(agg + (size_t)lds_dsti[wave * 2 + 1][rr] * 64 + col, v);
      }
  }
}

// ---------------------------------------------------------------------------
// Node kernel: per wave, TWO 16-node tiles, A = [agg | h] (16x128) -> h_out
// ---------------------------------------------------------------------------
__global__ __launch_bounds__(256) void gnn_node_kernel(
    const float* __restrict__ agg, const float* __restrict__ h_in,
    float* __restrict__ h_out,
    const __bf16* __restrict__ w0, const __bf16* __restrict__ w1,
    const __bf16* __restrict__ w2, const __bf16* __restrict__ w3,
    const float* __restrict__ b0, const float* __restrict__ b1,
    const float* __restrict__ b2, const float* __restrict__ b3,
    int n_tiles) {
  __shared__ __align__(16) __bf16 lds_tiles[16][16 * LDS_STRIDE];
  const int wave = threadIdx.x >> 5;
  const int lane = threadIdx.x & 31;
  const int t0 = (blockIdx.x * 8 + wave) * 2;
  if (t0 >= n_tiles) return;
  const bool has1 = (t0 + 1) < n_tiles;
  const int t1 = has1 ? (t0 + 1) : t0;
  __bf16* tile0 = &lds_tiles[wave * 2 + 0][0];
  __bf16* tile1 = &lds_tiles[wave * 2 + 1][0];
  const int row = lane & 15;
  const int khalf = lane >> 4;
  const int ni0 = t0 * 16 + row;
  const int ni1 = t1 * 16 + row;

  v8f acc0[4], acc1[4];
  init_bias(acc0, b0, row);
  init_bias(acc1, b0, row);
#pragma unroll
  for (int kb = 0; kb < 4; kb++) {
    v16bf a0, a1;
#pragma unroll
    for (int g = 0; g < 2; g++) {
      const int kbase = kb * 32 + g * 16;
      const int seg = kbase >> 6;              // 0: agg, 1: h
      const int off = (kbase & 63) + khalf * 8;
      const float* p0 = (seg == 0) ? (agg + (size_t)ni0 * 64 + off)
                                   : (h_in + (size_t)ni0 * 64 + off);
      const float* p1 = (seg == 0) ? (agg + (size_t)ni1 * 64 + off)
                                   : (h_in + (size_t)ni1 * 64 + off);
      cvt8(a0, g, p0);
      cvt8(a1, g, p1);
    }
    v16bf bf[4];
#pragma unroll
    for (int ct = 0; ct < 4; ct++)
      bf[ct] = *(const v16bf*)(w0 + ((size_t)(kb * 4 + ct) * 32 + lane) * 16);
#pragma unroll
    for (int ct = 0; ct < 4; ct++) {
      acc0[ct] = wmma_bf16(a0, bf[ct], acc0[ct]);
      acc1[ct] = wmma_bf16(a1, bf[ct], acc1[ct]);
    }
  }
  relu_store_lds(tile0, acc0, lane);
  relu_store_lds(tile1, acc1, lane);
  lds_fence();
  lds_mm64_pair(tile0, tile1, w1, b1, lane, acc0, acc1);
  relu_store_lds(tile0, acc0, lane);
  relu_store_lds(tile1, acc1, lane);
  lds_fence();
  lds_mm64_pair(tile0, tile1, w2, b2, lane, acc0, acc1);
  relu_store_lds(tile0, acc0, lane);
  relu_store_lds(tile1, acc1, lane);
  lds_fence();
  lds_mm64_pair(tile0, tile1, w3, b3, lane, acc0, acc1);

#pragma unroll
  for (int ct = 0; ct < 4; ct++)
#pragma unroll
    for (int r = 0; r < 8; r++) {
      const int rr = khalf * 8 + r;
      const int col = ct * 16 + row;
      float v = acc0[ct][r];
      v = v > 0.f ? v : 0.f;
      h_out[(size_t)(t0 * 16 + rr) * 64 + col] = v;
    }
  if (has1) {
#pragma unroll
    for (int ct = 0; ct < 4; ct++)
#pragma unroll
      for (int r = 0; r < 8; r++) {
        const int rr = khalf * 8 + r;
        const int col = ct * 16 + row;
        float v = acc1[ct][r];
        v = v > 0.f ? v : 0.f;
        h_out[(size_t)(t1 * 16 + rr) * 64 + col] = v;
      }
  }
}

// ---------------------------------------------------------------------------
// Host launch
// ---------------------------------------------------------------------------
extern "C" void kernel_launch(void* const* d_in, const int* in_sizes, int n_in,
                              void* d_out, int out_size, void* d_ws, size_t ws_size,
                              hipStream_t stream) {
  const int N = 50000, E = 800000, L = 3;

  const float* node_h = (const float*)d_in[0];
  const float* edge_h = (const float*)d_in[1];
  const int*   src    = (const int*)d_in[2];
  const int*   dst    = (const int*)d_in[3];
  const float *eW[4], *eb[4], *nW[4], *nb[4];
  for (int i = 0; i < 4; i++) {
    eW[i] = (const float*)d_in[4 + i * 4 + 0];
    eb[i] = (const float*)d_in[4 + i * 4 + 1];
    nW[i] = (const float*)d_in[4 + i * 4 + 2];
    nb[i] = (const float*)d_in[4 + i * 4 + 3];
  }

  // workspace carve-up (e: 204.8MB, h: 12.8MB, agg: 12.8MB, weights: ~270KB)
  char* p = (char*)d_ws;
  auto carve = [&](size_t bytes) -> char* {
    char* r = p;
    p += (bytes + 255) & ~(size_t)255;
    return r;
  };
  float*  e_buf = (float*)carve((size_t)E * 64 * 4);
  float*  h_buf = (float*)carve((size_t)N * 64 * 4);
  float*  agg   = (float*)carve((size_t)N * 64 * 4);
  __bf16* wpack = (__bf16*)carve((size_t)(3 * (192 * 64 + 3 * 64 * 64) +
                                          3 * (128 * 64 + 3 * 64 * 64)) * 2);

  // pre-pack all weights into B-fragment order (bf16)
  const __bf16* ewp[3][4];
  const __bf16* nwp[3][4];
  size_t woff = 0;
  for (int l = 0; l < L; l++) {
    for (int i = 0; i < 4; i++) {
      int Kin = (i == 0) ? 192 : 64;
      int total = Kin * 64;
      pack_weight_kernel<<<(total + 255) / 256, 256, 0, stream>>>(
          eW[i] + (size_t)l * total, wpack + woff, Kin);
      ewp[l][i] = wpack + woff;
      woff += total;
    }
    for (int i = 0; i < 4; i++) {
      int Kin = (i == 0) ? 128 : 64;
      int total = Kin * 64;
      pack_weight_kernel<<<(total + 255) / 256, 256, 0, stream>>>(
          nW[i] + (size_t)l * total, wpack + woff, Kin);
      nwp[l][i] = wpack + woff;
      woff += total;
    }
  }

  const int e_tiles = E / 16;              // 50000
  const int n_tiles = N / 16;              // 3125
  const int e_blocks = (e_tiles / 2 + 7) / 8;       // 2 tiles/wave, 8 waves/block
  const int n_blocks = ((n_tiles + 1) / 2 + 7) / 8;
  for (int l = 0; l < L; l++) {
    zero_f32_kernel<<<(N * 64 + 255) / 256, 256, 0, stream>>>(agg, N * 64);
    const float* h_in = (l == 0) ? node_h : h_buf;
    const float* e_in = (l == 0) ? edge_h : e_buf;
    gnn_edge_kernel<<<e_blocks, 256, 0, stream>>>(
        h_in, e_in, e_buf, agg, src, dst,
        ewp[l][0], ewp[l][1], ewp[l][2], ewp[l][3],
        eb[0] + l * 64, eb[1] + l * 64, eb[2] + l * 64, eb[3] + l * 64, e_tiles);
    float* h_out = (l == L - 1) ? (float*)d_out : h_buf;
    gnn_node_kernel<<<n_blocks, 256, 0, stream>>>(
        agg, h_in, h_out,
        nwp[l][0], nwp[l][1], nwp[l][2], nwp[l][3],
        nb[0] + l * 64, nb[1] + l * 64, nb[2] + l * 64, nb[3] + l * 64, n_tiles);
  }
}
